// MultiheadSelfAttention_1821066133889
// MI455X (gfx1250) — compile-verified, matrix operand order checked by reading, not run on hardware
//
#include <hip/hip_runtime.h>

// ---------------- problem constants ----------------
#define B_   2
#define S_   2048
#define D_   1024
#define H_   16
#define DK_  64
#define NT_  (B_ * S_)          // 4096 tokens

// ---------------- WMMA vector types ----------------
typedef _Float16 v16h __attribute__((ext_vector_type(16)));
typedef _Float16 v8h  __attribute__((ext_vector_type(8)));
typedef float    v8f  __attribute__((ext_vector_type(8)));
typedef int      v4i  __attribute__((ext_vector_type(4)));

// ---------------- CDNA5 async global->LDS copy (ASYNCcnt path) ----------------
#if defined(__has_builtin)
#if __has_builtin(__builtin_amdgcn_global_load_async_to_lds_b128) && \
    __has_builtin(__builtin_amdgcn_s_wait_asynccnt)
#define ASYNC_LDS 1
#endif
#endif
#ifndef ASYNC_LDS
#define ASYNC_LDS 0
#endif

#if ASYNC_LDS
typedef __attribute__((address_space(1))) void  gas_void;
typedef __attribute__((address_space(3))) void  las_void;
typedef __attribute__((address_space(1))) v4i   gas_v4i;
typedef __attribute__((address_space(3))) v4i   las_v4i;

__device__ __forceinline__ void async_cp16(_Float16* l, const _Float16* g) {
    // per-lane 16B copy, global -> LDS, no VGPR round trip (ASYNCcnt tracked)
    gas_v4i* gp = (gas_v4i*)(gas_void*)(void*)const_cast<_Float16*>(g);
    las_v4i* lp = (las_v4i*)(las_void*)(void*)l;
    __builtin_amdgcn_global_load_async_to_lds_b128(gp, lp, 0, 0);
}
#endif

// Build a 16-half A/B fragment for v_wmma_f32_16x16x32_f16.
// Per-lane layout (wave32): elements 0..7 = K k0..k0+7, elements 8..15 =
// K k0+16..k0+23, where the caller passes p0 = rowbase + k0 and
// k0 = (lane<16 ? 0 : 8).  Two contiguous 16-byte loads.
__device__ __forceinline__ v16h frag_ld(const _Float16* p0) {
    v8h lo = *(const v8h*)(p0);
    v8h hi = *(const v8h*)(p0 + 16);
    v16h r;
#pragma unroll
    for (int i = 0; i < 8; ++i) { r[i] = lo[i]; r[i + 8] = hi[i]; }
    return r;
}

// ---------------- fp32 -> f16 convert ----------------
__global__ void cvt_f32_f16(const float* __restrict__ src,
                            _Float16* __restrict__ dst, int n) {
    int i = blockIdx.x * blockDim.x + threadIdx.x;
    if (i < n) dst[i] = (_Float16)src[i];
}

// ---------------- tiled WMMA GEMM:  C[MxN] = A[MxK] * B[NxK]^T ----------------
// Block: 256 threads = 8 waves.  Block tile 128x64, wave tile 16x64, K-step 32.
// LDS tiles double-buffered; staged with async global->LDS copies when available.
#define GT_M 128
#define GT_N 64
#define GT_K 32
#define LSTR 40   // LDS row stride in halves (80B) -> conflict-free 16B lane reads

__global__ void __launch_bounds__(256)
gemm_f16_nt(const _Float16* __restrict__ A, const _Float16* __restrict__ Bw,
            float* __restrict__ C, int M, int N, int K) {
#if ASYNC_LDS
    __shared__ _Float16 lA[2][GT_M * LSTR];
    __shared__ _Float16 lB[2][GT_N * LSTR];
#else
    __shared__ _Float16 lA[1][GT_M * LSTR];
    __shared__ _Float16 lB[1][GT_N * LSTR];
#endif

    const int tid   = threadIdx.x;
    const int lane  = tid & 31;
    const int wave  = tid >> 5;
    const int m0    = blockIdx.y * GT_M;
    const int n0    = blockIdx.x * GT_N;
    const int lidx  = lane & 15;
    const int lhalf = (lane < 16) ? 0 : 8;
    const int rup   = (lane >> 4) << 3;

    v8f acc[4];
#pragma unroll
    for (int j = 0; j < 4; ++j)
#pragma unroll
        for (int r = 0; r < 8; ++r) acc[j][r] = 0.0f;

    // per-thread staging chunk: row = tid/4, 8-half column group = (tid%4)*8
    const int crow = tid >> 2;
    const int cq   = (tid & 3) * 8;
    const _Float16* gA = &A[(size_t)(m0 + crow) * K + cq];          // rows 0..63
    const _Float16* gB = &Bw[(size_t)(n0 + crow) * K + cq];

#if ASYNC_LDS
    // prologue: stage tile 0 asynchronously
    async_cp16(&lA[0][crow * LSTR + cq], gA);
    async_cp16(&lA[0][(crow + 64) * LSTR + cq], gA + (size_t)64 * K);
    async_cp16(&lB[0][crow * LSTR + cq], gB);

    for (int k0 = 0; k0 < K; k0 += GT_K) {
        const int  cur = (k0 >> 5) & 1;
        const bool has_next = (k0 + GT_K) < K;
        if (has_next) {
            const _Float16* gA2 = gA + k0 + GT_K;
            const _Float16* gB2 = gB + k0 + GT_K;
            async_cp16(&lA[cur ^ 1][crow * LSTR + cq], gA2);
            async_cp16(&lA[cur ^ 1][(crow + 64) * LSTR + cq], gA2 + (size_t)64 * K);
            async_cp16(&lB[cur ^ 1][crow * LSTR + cq], gB2);
            __builtin_amdgcn_s_wait_asynccnt(3);  // tile k0 done; next in flight
        } else {
            __builtin_amdgcn_s_wait_asynccnt(0);  // final tile done
        }
        __syncthreads();

        v16h af = frag_ld(&lA[cur][(wave * 16 + lidx) * LSTR + lhalf]);
#pragma unroll
        for (int j = 0; j < 4; ++j) {
            v16h bf = frag_ld(&lB[cur][(j * 16 + lidx) * LSTR + lhalf]);
            acc[j] = __builtin_amdgcn_wmma_f32_16x16x32_f16(
                false, af, false, bf, (short)0, acc[j], false, false);
        }
        __syncthreads();   // safe to overwrite buffer `cur` next iteration
    }
#else
    for (int k0 = 0; k0 < K; k0 += GT_K) {
        *(v8h*)&lA[0][crow * LSTR + cq] = *(const v8h*)(gA + k0);
        *(v8h*)&lA[0][(crow + 64) * LSTR + cq] = *(const v8h*)(gA + (size_t)64 * K + k0);
        *(v8h*)&lB[0][crow * LSTR + cq] = *(const v8h*)(gB + k0);
        __syncthreads();

        v16h af = frag_ld(&lA[0][(wave * 16 + lidx) * LSTR + lhalf]);
#pragma unroll
        for (int j = 0; j < 4; ++j) {
            v16h bf = frag_ld(&lB[0][(j * 16 + lidx) * LSTR + lhalf]);
            acc[j] = __builtin_amdgcn_wmma_f32_16x16x32_f16(
                false, af, false, bf, (short)0, acc[j], false, false);
        }
        __syncthreads();
    }
#endif

    const int rbase = m0 + wave * 16 + rup;
#pragma unroll
    for (int j = 0; j < 4; ++j)
#pragma unroll
        for (int r = 0; r < 8; ++r)
            C[(size_t)(rbase + r) * N + n0 + j * 16 + lidx] = acc[j][r];
}

// ---------------- RoPE + repack Q,K : fp32 [n][D] -> f16 [B][H][S][64] ----------------
__global__ void rope_pack(const float* __restrict__ qf, const float* __restrict__ kf,
                          const int* __restrict__ pos,
                          _Float16* __restrict__ qb, _Float16* __restrict__ kb) {
    int id = blockIdx.x * blockDim.x + threadIdx.x;     // one pair per thread
    if (id >= NT_ * D_ / 2) return;
    const int ppt = D_ / 2;                // 512 pairs per token
    int n = id / ppt;
    int p = id - n * ppt;
    int h = p >> 5;                        // head
    int i = p & 31;                        // pair index within head
    int b = n / S_;
    int s = n - b * S_;

    float ang = (float)pos[s] * __powf(10000.0f, -((float)(2 * i) / (float)DK_));
    float sn, cs;
    __sincosf(ang, &sn, &cs);

    int    src = n * D_ + h * DK_ + 2 * i;
    size_t dst = ((size_t)(b * H_ + h) * S_ + s) * DK_ + 2 * i;

    float q1 = qf[src], q2 = qf[src + 1];
    qb[dst]     = (_Float16)(q1 * cs - q2 * sn);
    qb[dst + 1] = (_Float16)(q1 * sn + q2 * cs);
    float k1 = kf[src], k2 = kf[src + 1];
    kb[dst]     = (_Float16)(k1 * cs - k2 * sn);
    kb[dst + 1] = (_Float16)(k1 * sn + k2 * cs);
}

// ---------------- V repack transposed : fp32 [n][D] -> f16 [B][H][64][S] ----------------
__global__ void v_trans(const float* __restrict__ vf, _Float16* __restrict__ vt) {
    int id = blockIdx.x * blockDim.x + threadIdx.x;
    if (id >= NT_ * D_) return;
    int n = id / D_;
    int c = id - n * D_;
    int h = c >> 6, d = c & 63;
    int b = n / S_, s = n - b * S_;
    vt[((size_t)(b * H_ + h) * DK_ + d) * S_ + s] = (_Float16)vf[id];
}

// ---------------- flash attention : one wave per 16-query tile ----------------
__global__ void __launch_bounds__(32)
flash_attn(const _Float16* __restrict__ qb, const _Float16* __restrict__ kb,
           const _Float16* __restrict__ vt, _Float16* __restrict__ hb) {
    __shared__ _Float16 pl[16 * LSTR];     // P tile bounce buffer (C-layout -> A-layout)

    const int lane  = threadIdx.x;
    const int qt    = blockIdx.x;          // 0..127
    const int h     = blockIdx.y;
    const int b     = blockIdx.z;
    const int q0    = qt * 16;
    const int bh    = b * H_ + h;
    const int lidx  = lane & 15;
    const int lhalf = (lane < 16) ? 0 : 8;
    const int rup   = (lane >> 4) << 3;

    // Q fragments (A-layout), dk 0..31 and 32..63 — loaded once
    const _Float16* qrow = qb + ((size_t)bh * S_ + q0 + lidx) * DK_;
    v16h qf0 = frag_ld(qrow + lhalf);
    v16h qf1 = frag_ld(qrow + 32 + lhalf);

    v8f acc[4];
    float mrun[8], lrun[8];
#pragma unroll
    for (int r = 0; r < 8; ++r) { mrun[r] = -1e30f; lrun[r] = 0.0f; }
#pragma unroll
    for (int j = 0; j < 4; ++j)
#pragma unroll
        for (int r = 0; r < 8; ++r) acc[j][r] = 0.0f;

    const int nblk = (q0 + 16 + 31) >> 5;  // 32-key blocks covering causal range
    for (int kbk = 0; kbk < nblk; ++kbk) {
        const int key0 = kbk * 32;

        // ---- scores: Q(16x64) . K^T(64x32) as two 16x16 tiles, 2 WMMA each
        v8f s0, s1;
#pragma unroll
        for (int r = 0; r < 8; ++r) { s0[r] = 0.0f; s1[r] = 0.0f; }
        {
            const _Float16* kr0 = kb + ((size_t)bh * S_ + key0 + lidx) * DK_;
            v16h kf0 = frag_ld(kr0 + lhalf);
            v16h kf1 = frag_ld(kr0 + 32 + lhalf);
            s0 = __builtin_amdgcn_wmma_f32_16x16x32_f16(false, qf0, false, kf0, (short)0, s0, false, false);
            s0 = __builtin_amdgcn_wmma_f32_16x16x32_f16(false, qf1, false, kf1, (short)0, s0, false, false);
            const _Float16* kr1 = kr0 + (size_t)16 * DK_;
            v16h kg0 = frag_ld(kr1 + lhalf);
            v16h kg1 = frag_ld(kr1 + 32 + lhalf);
            s1 = __builtin_amdgcn_wmma_f32_16x16x32_f16(false, qf0, false, kg0, (short)0, s1, false, false);
            s1 = __builtin_amdgcn_wmma_f32_16x16x32_f16(false, qf1, false, kg1, (short)0, s1, false, false);
        }

        // ---- scale + causal mask + streaming softmax
        const float scale = 0.125f;        // 1/sqrt(64)
        float tmax[8];
#pragma unroll
        for (int r = 0; r < 8; ++r) {
            int qg = q0 + r + rup;
            float a = s0[r] * scale; if (key0 + lidx      > qg) a = -1e30f;
            float c = s1[r] * scale; if (key0 + 16 + lidx > qg) c = -1e30f;
            s0[r] = a; s1[r] = c;
            tmax[r] = fmaxf(a, c);
        }
#pragma unroll
        for (int r = 0; r < 8; ++r) {      // row-max across 16-lane column group
            tmax[r] = fmaxf(tmax[r], __shfl_xor(tmax[r], 1));
            tmax[r] = fmaxf(tmax[r], __shfl_xor(tmax[r], 2));
            tmax[r] = fmaxf(tmax[r], __shfl_xor(tmax[r], 4));
            tmax[r] = fmaxf(tmax[r], __shfl_xor(tmax[r], 8));
        }
        float alpha[8], rs[8];
#pragma unroll
        for (int r = 0; r < 8; ++r) {
            float mnew = fmaxf(mrun[r], tmax[r]);
            alpha[r] = __expf(mrun[r] - mnew);
            mrun[r] = mnew;
            s0[r] = __expf(s0[r] - mnew);
            s1[r] = __expf(s1[r] - mnew);
            rs[r] = s0[r] + s1[r];
        }
#pragma unroll
        for (int r = 0; r < 8; ++r) {      // row-sum across 16-lane column group
            rs[r] += __shfl_xor(rs[r], 1);
            rs[r] += __shfl_xor(rs[r], 2);
            rs[r] += __shfl_xor(rs[r], 4);
            rs[r] += __shfl_xor(rs[r], 8);
            lrun[r] = lrun[r] * alpha[r] + rs[r];
        }
#pragma unroll
        for (int j = 0; j < 4; ++j)
#pragma unroll
            for (int r = 0; r < 8; ++r) acc[j][r] *= alpha[r];

        // ---- P: C-layout -> A-layout via LDS
#pragma unroll
        for (int r = 0; r < 8; ++r) {
            int row = r + rup;
            pl[row * LSTR + lidx]      = (_Float16)s0[r];
            pl[row * LSTR + 16 + lidx] = (_Float16)s1[r];
        }
        __syncthreads();
        v16h pf = frag_ld(&pl[lidx * LSTR + lhalf]);
        __syncthreads();

        // ---- acc += P(16x32) . V(32x64): 4 dim-tiles, 1 WMMA each
#pragma unroll
        for (int j = 0; j < 4; ++j) {
            const _Float16* vr = vt + ((size_t)bh * DK_ + j * 16 + lidx) * S_ + key0;
            v16h vfr = frag_ld(vr + lhalf);
            acc[j] = __builtin_amdgcn_wmma_f32_16x16x32_f16(
                false, pf, false, vfr, (short)0, acc[j], false, false);
        }
    }

    // ---- epilogue: normalize, write heads output [n][D] (f16 for O-proj GEMM)
#pragma unroll
    for (int r = 0; r < 8; ++r) {
        float inv = 1.0f / lrun[r];
        size_t rowo = ((size_t)b * S_ + q0 + r + rup) * D_ + h * DK_;
#pragma unroll
        for (int j = 0; j < 4; ++j)
            hb[rowo + j * 16 + lidx] = (_Float16)(acc[j][r] * inv);
    }
}

// ---------------- host-side orchestration ----------------
extern "C" void kernel_launch(void* const* d_in, const int* in_sizes, int n_in,
                              void* d_out, int out_size, void* d_ws, size_t ws_size,
                              hipStream_t stream) {
    (void)in_sizes; (void)n_in; (void)out_size; (void)ws_size;

    const float* x   = (const float*)d_in[0];
    const int*   pos = (const int*)d_in[1];
    const float* wq  = (const float*)d_in[2];
    const float* wk  = (const float*)d_in[3];
    const float* wv  = (const float*)d_in[4];
    const float* wo  = (const float*)d_in[5];

    char* ws = (char*)d_ws;
    const size_t MiB = 1024u * 1024u;
    _Float16* xb  = (_Float16*)(ws + 0 * MiB);    // 8 MiB  x in f16
    _Float16* wqb = (_Float16*)(ws + 8 * MiB);    // 2 MiB
    _Float16* wkb = (_Float16*)(ws + 10 * MiB);   // 2 MiB
    _Float16* wvb = (_Float16*)(ws + 12 * MiB);   // 2 MiB
    _Float16* wob = (_Float16*)(ws + 14 * MiB);   // 2 MiB
    float*    qf  = (float*)   (ws + 16 * MiB);   // 16 MiB Q fp32 pre-RoPE
    float*    kf  = (float*)   (ws + 32 * MiB);   // 16 MiB
    float*    vf  = (float*)   (ws + 48 * MiB);   // 16 MiB
    _Float16* qb  = (_Float16*)(ws + 64 * MiB);   // 8 MiB  [B][H][S][64]
    _Float16* kb  = (_Float16*)(ws + 72 * MiB);   // 8 MiB  [B][H][S][64]
    _Float16* vt  = (_Float16*)(ws + 80 * MiB);   // 8 MiB  [B][H][64][S]
    _Float16* hb  = (_Float16*)(ws + 88 * MiB);   // 8 MiB  heads out [n][D]

    // 1) fp32 -> f16
    cvt_f32_f16<<<(NT_ * D_ + 255) / 256, 256, 0, stream>>>(x, xb, NT_ * D_);
    cvt_f32_f16<<<(D_ * D_ + 255) / 256, 256, 0, stream>>>(wq, wqb, D_ * D_);
    cvt_f32_f16<<<(D_ * D_ + 255) / 256, 256, 0, stream>>>(wk, wkb, D_ * D_);
    cvt_f32_f16<<<(D_ * D_ + 255) / 256, 256, 0, stream>>>(wv, wvb, D_ * D_);
    cvt_f32_f16<<<(D_ * D_ + 255) / 256, 256, 0, stream>>>(wo, wob, D_ * D_);

    // 2) Q/K/V projections (C = X . W^T)
    dim3 gg(D_ / GT_N, NT_ / GT_M);
    gemm_f16_nt<<<gg, 256, 0, stream>>>(xb, wqb, qf, NT_, D_, D_);
    gemm_f16_nt<<<gg, 256, 0, stream>>>(xb, wkb, kf, NT_, D_, D_);
    gemm_f16_nt<<<gg, 256, 0, stream>>>(xb, wvb, vf, NT_, D_, D_);

    // 3) RoPE + repack, V transpose-repack
    rope_pack<<<(NT_ * D_ / 2 + 255) / 256, 256, 0, stream>>>(qf, kf, pos, qb, kb);
    v_trans<<<(NT_ * D_ + 255) / 256, 256, 0, stream>>>(vf, vt);

    // 4) causal flash attention
    flash_attn<<<dim3(S_ / 16, H_, B_), 32, 0, stream>>>(qb, kb, vt, hb);

    // 5) output projection straight into d_out (fp32)
    gemm_f16_nt<<<gg, 256, 0, stream>>>(hb, wob, (float*)d_out, NT_, D_, D_);
}
